// OneHopThenKConv_27590869909895
// MI455X (gfx1250) — compile-verified, matrix operand order checked by reading, not run on hardware
//
#include <hip/hip_runtime.h>

typedef __attribute__((ext_vector_type(16))) _Float16 v16h;
typedef __attribute__((ext_vector_type(8)))  _Float16 v8h;
typedef __attribute__((ext_vector_type(8)))  float    v8f;
typedef __attribute__((ext_vector_type(4)))  unsigned v4u;
typedef __attribute__((ext_vector_type(8)))  int      v8i;
typedef __attribute__((ext_vector_type(4)))  int      v4i;

#define N_NODES 50000
#define N_EDGES 800000
#define HDIM 128

// ---------------------------------------------------------------------------
// helpers
// ---------------------------------------------------------------------------
__device__ __forceinline__ unsigned pack2h(float a, float b) {
    union { _Float16 h[2]; unsigned u; } r;
    r.h[0] = (_Float16)a; r.h[1] = (_Float16)b;
    return r.u;
}

// A-fragment (16x32 f16, MxK) from an LDS row-major tile [16][stride] halfs.
// ISA layout: lane L -> M = L%16; element e -> K = (e<8 ? e : e+8) + 8*(L/16).
__device__ __forceinline__ v16h load_afrag(const _Float16* lds, int lane, int kt, int stride) {
    const int row = lane & 15;
    const int hh  = (lane >> 4) << 3;
    const unsigned short* p = (const unsigned short*)(lds + row * stride);
    union { v16h v; unsigned u[8]; } r;
#pragma unroll
    for (int i = 0; i < 8; ++i) {
        int e = 2 * i;
        int k = ((e < 8) ? e : e + 8) + hh + (kt << 5);
        r.u[i] = *(const unsigned*)(p + k);   // k is even -> 4B aligned
    }
    return r.v;
}

// A-fragment straight from a global f16 row: two contiguous 16B loads.
__device__ __forceinline__ v16h load_afrag_row(const _Float16* row, int c0) {
    union { v16h v; v8h h[2]; } a;
    a.h[0] = *(const v8h*)(row + c0);
    a.h[1] = *(const v8h*)(row + c0 + 16);
    return a.v;
}

// B-fragment: weights pre-shuffled into fragment-major layout, so each lane
// just loads 16 contiguous halfs (32B) -> coalesced / bank-conflict-free.
__device__ __forceinline__ v16h load_bfrag(const _Float16* Wf, int frag, int lane) {
    return *(const v16h*)(Wf + ((size_t)frag * 32 + lane) * 16);
}

// ---------------------------------------------------------------------------
// weight conversion: f32 row-major [Kdim][128] -> f16 WMMA B-fragment layout
// ---------------------------------------------------------------------------
__global__ void convert_weight_kernel(const float* __restrict__ W,
                                      _Float16* __restrict__ Wf, int Kdim) {
    int tid = blockIdx.x * blockDim.x + threadIdx.x;
    if (tid >= Kdim * HDIM) return;
    int e = tid & 15;
    int lane = (tid >> 4) & 31;
    int f = tid >> 9;
    int kt = f >> 3, nt = f & 7;
    int k = ((e < 8) ? e : e + 8) + ((lane >> 4) << 3) + (kt << 5);
    int n = (nt << 4) + (lane & 15);
    Wf[tid] = (_Float16)W[k * HDIM + n];
}

// x f32 -> f16 row-major (done once; halves all edge-gather traffic)
__global__ void convert_x_kernel(const float* __restrict__ x, _Float16* __restrict__ xh) {
    int i = blockIdx.x * blockDim.x + threadIdx.x;  // one float4 per thread
    if (i >= (N_NODES * HDIM) / 4) return;
    float4 v = *(const float4*)(x + (size_t)i * 4);
    unsigned* o = (unsigned*)(xh + (size_t)i * 4);
    o[0] = pack2h(v.x, v.y);
    o[1] = pack2h(v.z, v.w);
}

// ---------------------------------------------------------------------------
// gcn_norm pre-pass
// ---------------------------------------------------------------------------
__global__ void degree_kernel(const int* __restrict__ dst, float* __restrict__ deg) {
    int e = blockIdx.x * blockDim.x + threadIdx.x;
    if (e < N_EDGES) atomicAdd(&deg[dst[e]], 1.0f);
}

__global__ void dinv_kernel(float* __restrict__ deg) {
    int i = blockIdx.x * blockDim.x + threadIdx.x;
    if (i < N_NODES) {
        float d = deg[i];
        deg[i] = (d > 0.0f) ? rsqrtf(fmaxf(d, 1e-12f)) : 0.0f;
    }
}

__global__ void norm_kernel(const int* __restrict__ src, const int* __restrict__ dst,
                            const float* __restrict__ dinv, float* __restrict__ nrm) {
    int e = blockIdx.x * blockDim.x + threadIdx.x;
    if (e < N_EDGES) nrm[e] = dinv[src[e]] * dinv[dst[e]];
}

// ---------------------------------------------------------------------------
// Edge MLP: m = (LeakyReLU([x_dst|x_src]W1+b1))W2+b2, scatter-add @dst.
// 8 waves x 16 edges; A-fragments gathered directly from f16 x rows.
// ---------------------------------------------------------------------------
__global__ void __launch_bounds__(256) mp_edge_kernel(
    const _Float16* __restrict__ xh, const int* __restrict__ src, const int* __restrict__ dst,
    const _Float16* __restrict__ W1f, const float* __restrict__ b1,
    const _Float16* __restrict__ W2f, const float* __restrict__ b2,
    float* __restrict__ out)
{
    constexpr int SH = 136;   // 128 + 8 pad halfs
    __shared__ _Float16 ldsH[8][16 * SH];   // 34816 B

    const int lane  = threadIdx.x & 31;
    const int wave  = threadIdx.x >> 5;
    const int ebase = blockIdx.x * 128 + wave * 16;
    const int m0    = (lane >> 4) << 3;     // also the K-subblock shift (8h)
    const int n0    = lane & 15;

    // each lane owns matrix-row (edge) ebase + lane%16
    const int erow = ebase + n0;
    const _Float16* rowd = xh + (size_t)dst[erow] * HDIM;
    const _Float16* rows = xh + (size_t)src[erow] * HDIM;
    _Float16* Hh = ldsH[wave];

    // GEMM1: [16x256] @ [256x128] + b1, LeakyReLU(0.2)
    v8f acc[8];
#pragma unroll
    for (int nt = 0; nt < 8; ++nt) {
        float bv = b1[nt * 16 + n0];
#pragma unroll
        for (int r = 0; r < 8; ++r) acc[nt][r] = bv;
    }
#pragma unroll 1
    for (int kt = 0; kt < 8; ++kt) {
        const _Float16* pr = (kt < 4) ? rowd : rows;   // concat [x_dst | x_src]
        v16h a = load_afrag_row(pr, ((kt & 3) << 5) + m0);
#pragma unroll
        for (int nt = 0; nt < 8; ++nt) {
            v16h b = load_bfrag(W1f, kt * 8 + nt, lane);
            acc[nt] = __builtin_amdgcn_wmma_f32_16x16x32_f16(
                false, a, false, b, (short)0, acc[nt], false, false);
        }
    }
#pragma unroll
    for (int nt = 0; nt < 8; ++nt)
#pragma unroll
        for (int r = 0; r < 8; ++r) {
            float v = acc[nt][r];
            v = (v > 0.0f) ? v : 0.2f * v;
            Hh[(m0 + r) * SH + nt * 16 + n0] = (_Float16)v;   // transpose via LDS
        }
    __syncthreads();

    // GEMM2: [16x128] @ [128x128] + b2 -> scatter atomic add at dst
#pragma unroll
    for (int nt = 0; nt < 8; ++nt) {
        float bv = b2[nt * 16 + n0];
#pragma unroll
        for (int r = 0; r < 8; ++r) acc[nt][r] = bv;
    }
#pragma unroll 1
    for (int kt = 0; kt < 4; ++kt) {
        v16h a = load_afrag(Hh, lane, kt, SH);
#pragma unroll
        for (int nt = 0; nt < 8; ++nt) {
            v16h b = load_bfrag(W2f, kt * 8 + nt, lane);
            acc[nt] = __builtin_amdgcn_wmma_f32_16x16x32_f16(
                false, a, false, b, (short)0, acc[nt], false, false);
        }
    }
#pragma unroll
    for (int r = 0; r < 8; ++r) {
        int e = ebase + m0 + r;
        float* orow = out + (size_t)dst[e] * HDIM + n0;
#pragma unroll
        for (int nt = 0; nt < 8; ++nt) atomicAdd(orow + nt * 16, acc[nt][r]);
    }
}

// ---------------------------------------------------------------------------
// dense node GEMM: Cout = op(A[nrows x128] @ W[128x128])
// Weight fragment block (32KB) is DMA'd into LDS once per block by the
// Tensor Data Mover; B-fragments then come from LDS.
// mode bit0: init with bias (else accumulate Cin); bit1: ReLU output
// ---------------------------------------------------------------------------
__global__ void __launch_bounds__(128) gemm_node_kernel(
    const float* __restrict__ A, const _Float16* __restrict__ Wf,
    const float* __restrict__ Cin, float* __restrict__ Cout,
    const float* __restrict__ bias, int mode, int nrows)
{
    constexpr int SA = 136;
    __shared__ _Float16 ldsA[4][16 * SA];          // 17408 B
    __shared__ _Float16 ldsW[32 * 32 * 16];        // 32768 B  (32 fragments)

    const int lane  = threadIdx.x & 31;
    const int wave  = threadIdx.x >> 5;
    const int rbase = blockIdx.x * 64 + wave * 16;
    _Float16* Al = ldsA[wave];
    unsigned* A32 = (unsigned*)Al;

    if (wave == 0) {
        // TDM descriptor: 1 row of 4096 x 8B elements (= 32KB), global -> LDS
        uintptr_t ga = (uintptr_t)Wf;
        unsigned   la = (unsigned)(uintptr_t)(&ldsW[0]);
        v4u g0 = { 1u,                                    // count=1, user desc
                   la,                                    // lds_addr
                   (unsigned)(ga & 0xFFFFFFFFu),          // global_addr[31:0]
                   (unsigned)(((ga >> 32) & 0x01FFFFFFu) | 0x80000000u) }; // addr[56:32] | type=2
        const int td0 = 4096;                             // elements of 8B
        v8i g1 = { (int)(3u << 16),                       // data_size = 8B
                   (int)((td0 & 0xFFFF) << 16),           // tensor_dim0 lo16
                   (int)(1 << 16),                        // dim0 hi=0 | tensor_dim1 lo16=1
                   (int)(td0 << 16),                      // dim1 hi=0 | tile_dim0=4096
                   1,                                     // tile_dim1=1, tile_dim2=0
                   td0,                                   // tensor_dim0_stride lo32
                   0, 0 };
        v4i g2 = { 0, 0, 0, 0 };
        v4i g3 = { 0, 0, 0, 0 };
        v8i g4 = { 0, 0, 0, 0, 0, 0, 0, 0 };
        __builtin_amdgcn_tensor_load_to_lds(g0, g1, g2, g3, g4, 0);
        __builtin_amdgcn_s_wait_tensorcnt(0);
    }

    // stage + convert A rows to f16 in LDS
#pragma unroll 1
    for (int r = 0; r < 16; ++r) {
        int row = rbase + r;
        float4 xv = make_float4(0.f, 0.f, 0.f, 0.f);
        if (row < nrows) xv = *(const float4*)(A + (size_t)row * HDIM + lane * 4);
        int base = r * SA + lane * 4;
        A32[(base >> 1) + 0] = pack2h(xv.x, xv.y);
        A32[(base >> 1) + 1] = pack2h(xv.z, xv.w);
    }
    __syncthreads();

    const int m0 = (lane >> 4) << 3;
    const int n0 = lane & 15;
    v8f acc[8];
#pragma unroll
    for (int nt = 0; nt < 8; ++nt) {
        float bv = (mode & 1) ? bias[nt * 16 + n0] : 0.0f;
#pragma unroll
        for (int r = 0; r < 8; ++r) acc[nt][r] = bv;
    }
#pragma unroll 1
    for (int kt = 0; kt < 4; ++kt) {
        v16h a = load_afrag(Al, lane, kt, SA);
#pragma unroll
        for (int nt = 0; nt < 8; ++nt) {
            v16h b = load_bfrag(ldsW, kt * 8 + nt, lane);   // ds_load_b128 x2
            acc[nt] = __builtin_amdgcn_wmma_f32_16x16x32_f16(
                false, a, false, b, (short)0, acc[nt], false, false);
        }
    }
#pragma unroll
    for (int r = 0; r < 8; ++r) {
        int row = rbase + m0 + r;
        if (row >= nrows) continue;
#pragma unroll
        for (int nt = 0; nt < 8; ++nt) {
            size_t idx = (size_t)row * HDIM + nt * 16 + n0;
            float v = acc[nt][r];
            if (!(mode & 1)) v += Cin[idx];
            if (mode & 2) v = fmaxf(v, 0.0f);
            Cout[idx] = v;
        }
    }
}

// ---------------------------------------------------------------------------
// SpMM propagation: y[dst] += norm[e] * x[src], one wave per edge
// ---------------------------------------------------------------------------
__global__ void spmm_kernel(const float* __restrict__ x, const int* __restrict__ src,
                            const int* __restrict__ dst, const float* __restrict__ nrm,
                            float* __restrict__ y)
{
    int gid = blockIdx.x * blockDim.x + threadIdx.x;
    int e = gid >> 5;
    int lane = gid & 31;
    if (e >= N_EDGES) return;
    float nm = nrm[e];
    float4 xv = *(const float4*)(x + (size_t)src[e] * HDIM + lane * 4);
    float* yr = y + (size_t)dst[e] * HDIM + lane * 4;
    atomicAdd(yr + 0, nm * xv.x);
    atomicAdd(yr + 1, nm * xv.y);
    atomicAdd(yr + 2, nm * xv.z);
    atomicAdd(yr + 3, nm * xv.w);
}

// ---------------------------------------------------------------------------
// host
// ---------------------------------------------------------------------------
extern "C" void kernel_launch(void* const* d_in, const int* in_sizes, int n_in,
                              void* d_out, int out_size, void* d_ws, size_t ws_size,
                              hipStream_t stream) {
    (void)in_sizes; (void)n_in; (void)out_size; (void)ws_size;

    const float* x  = (const float*)d_in[0];
    const int*   ei = (const int*)d_in[1];
    const float* W1 = (const float*)d_in[2];
    const float* b1 = (const float*)d_in[3];
    const float* W2 = (const float*)d_in[4];
    const float* b2 = (const float*)d_in[5];
    const float* tw = (const float*)d_in[6];
    const float* tb = (const float*)d_in[7];
    float* out = (float*)d_out;

    const int* src = ei;
    const int* dst = ei + N_EDGES;

    // workspace layout (256B aligned slices)
    char* p = (char*)d_ws;
    size_t off = 0;
    auto take = [&](size_t bytes) -> char* {
        char* r = p + off;
        off = (off + bytes + 255) & ~(size_t)255;
        return r;
    };
    _Float16* W1f  = (_Float16*)take((size_t)256 * HDIM * 2);
    _Float16* W2f  = (_Float16*)take((size_t)HDIM * HDIM * 2);
    _Float16* TAGf = (_Float16*)take((size_t)12 * HDIM * HDIM * 2);
    _Float16* xhh  = (_Float16*)take((size_t)N_NODES * HDIM * 2);
    float* dinvb = (float*)take((size_t)N_NODES * 4);
    float* normv = (float*)take((size_t)N_EDGES * 4);
    float* bufA  = (float*)take((size_t)N_NODES * HDIM * 4);
    float* bufB  = (float*)take((size_t)N_NODES * HDIM * 4);
    float* bufC  = (float*)take((size_t)N_NODES * HDIM * 4);

    const size_t NH_BYTES = (size_t)N_NODES * HDIM * 4;

    // weights -> f16 fragment layout; x -> f16 rows
    convert_weight_kernel<<<(256 * HDIM + 255) / 256, 256, 0, stream>>>(W1, W1f, 256);
    convert_weight_kernel<<<(HDIM * HDIM + 255) / 256, 256, 0, stream>>>(W2, W2f, HDIM);
    for (int i = 0; i < 12; ++i)
        convert_weight_kernel<<<(HDIM * HDIM + 255) / 256, 256, 0, stream>>>(
            tw + (size_t)i * HDIM * HDIM, TAGf + (size_t)i * HDIM * HDIM, HDIM);
    convert_x_kernel<<<(N_NODES * HDIM / 4 + 255) / 256, 256, 0, stream>>>(x, xhh);

    // gcn norm
    (void)hipMemsetAsync(dinvb, 0, (size_t)N_NODES * 4, stream);
    degree_kernel<<<(N_EDGES + 255) / 256, 256, 0, stream>>>(dst, dinvb);
    dinv_kernel<<<(N_NODES + 255) / 256, 256, 0, stream>>>(dinvb);
    norm_kernel<<<(N_EDGES + 255) / 256, 256, 0, stream>>>(src, dst, dinvb, normv);

    // edge MLP + scatter
    (void)hipMemsetAsync(bufA, 0, NH_BYTES, stream);
    mp_edge_kernel<<<N_EDGES / 128, 256, 0, stream>>>(xhh, src, dst, W1f, b1, W2f, b2, bufA);

    // TAGConv stack
    const int GB = (N_NODES + 63) / 64;
    float* cur  = bufA;
    float* accb = bufB;
    float* tmp  = bufC;
    for (int l = 0; l < 3; ++l) {
        const _Float16* Wl = TAGf + (size_t)l * 4 * HDIM * HDIM;
        const float* bl = tb + l * HDIM;
        // k = 0: acc = cur @ W0 + b
        gemm_node_kernel<<<GB, 128, 0, stream>>>(cur, Wl, cur, accb, bl, 1, N_NODES);
        float* hin = cur;
        float* hout = tmp;
        for (int k = 1; k <= 3; ++k) {
            (void)hipMemsetAsync(hout, 0, NH_BYTES, stream);
            spmm_kernel<<<(N_EDGES * 32 + 255) / 256, 256, 0, stream>>>(hin, src, dst, normv, hout);
            int last = (k == 3);
            float* dstbuf = (l == 2 && last) ? out : accb;
            gemm_node_kernel<<<GB, 128, 0, stream>>>(
                hout, Wl + (size_t)k * HDIM * HDIM, accb, dstbuf, bl, last ? 2 : 0, N_NODES);
            float* t = hin; hin = hout; hout = t;
        }
        if (l < 2) { float* t = cur; cur = accb; accb = t; }
    }
}
